// SelfAttention_40355512713594
// MI455X (gfx1250) — matrix operand order repaired and checked by reading, not compiled, not run on hardware
//
#include <hip/hip_runtime.h>
#include <hip/hip_bf16.h>

// ---------------------------------------------------------------------------
// Self-attention forward for MI455X (gfx1250, wave32, WMMA).
// B=2, S=2048, D=1024, H=16, DH=64.  All matmuls via v_wmma_f32_16x16x32_bf16.
// ---------------------------------------------------------------------------

#define B_   2
#define S_   2048
#define D_   1024
#define H_   16
#define DH_  64
#define MROWS (B_ * S_)   // 4096

// GEMM tiling: block = 256 threads (8 waves, 4x2), tile 128(M) x 64(N), BK=32.
// Each wave computes a 32x32 macro-tile = 4 WMMAs per K-step.
#define BM 128
#define BN 64
#define LSTRIDE 20   // uints per LDS row (80B): 16B-aligned, conflict-free

#define AS1 __attribute__((address_space(1)))
#define AS3 __attribute__((address_space(3)))

#if defined(__has_builtin)
#  if __has_builtin(__builtin_amdgcn_global_load_async_to_lds_b128)
#    define HAVE_ASYNC_LDS 1
#  endif
#endif

typedef __attribute__((ext_vector_type(16))) __bf16 v16bf;
typedef __attribute__((ext_vector_type(8)))  float  v8f;
typedef int v4i_vs __attribute__((vector_size(16)));   // matches builtin param type

union FragB16 {
    unsigned u[8];
    v16bf    v;
};

// round-to-nearest-even fp32 -> bf16
__device__ __forceinline__ unsigned short f2bf(float x) {
    unsigned a = __float_as_uint(x);
    a = (a + 0x7FFFu + ((a >> 16) & 1u)) >> 16;
    return (unsigned short)a;
}
__device__ __forceinline__ unsigned pack2(float lo, float hi) {
    unsigned a = __float_as_uint(lo);
    unsigned b = __float_as_uint(hi);
    a = (a + 0x7FFFu + ((a >> 16) & 1u)) >> 16;
    b = (b + 0x7FFFu + ((b >> 16) & 1u)) >> 16;
    return a | (b << 16);
}

// reductions within each 16-lane half of the wave32
__device__ __forceinline__ float hmax16(float x) {
    #pragma unroll
    for (int m = 1; m < 16; m <<= 1) x = fmaxf(x, __shfl_xor(x, m, 32));
    return x;
}
__device__ __forceinline__ float hsum16(float x) {
    #pragma unroll
    for (int m = 1; m < 16; m <<= 1) x += __shfl_xor(x, m, 32);
    return x;
}

// A-fragment K index for 16-bit 16x32 layout
__device__ __forceinline__ int afrag_k0(int v, int half) {
    return ((v >> 2) << 4) + ((v & 3) << 1) + (half << 3);
}

__device__ __forceinline__ void async_wait0() {
#if defined(HAVE_ASYNC_LDS)
#  if __has_builtin(__builtin_amdgcn_s_wait_asynccnt)
    __builtin_amdgcn_s_wait_asynccnt(0);
#  else
    asm volatile("s_wait_asynccnt 0" ::: "memory");
#  endif
#endif
}

// ---------------------------------------------------------------------------
// Kernel 1: fused QKV projection.  Y = x @ W^T + b  (fp32 in, bf16 out)
//   grid = (M/128, N/64, 3)   block = 256
//   z=0 -> Q [BH,S,DH]   z=1 -> K [BH,S,DH]   z=2 -> V^T [BH,DH,S]
// ---------------------------------------------------------------------------
__global__ __launch_bounds__(256) void qkv_kernel(
    const float* __restrict__ x,
    const float* __restrict__ Wq, const float* __restrict__ Wk, const float* __restrict__ Wv,
    const float* __restrict__ bq, const float* __restrict__ bk, const float* __restrict__ bv,
    unsigned short* __restrict__ qo, unsigned short* __restrict__ ko,
    unsigned short* __restrict__ vo)
{
    const int mode = blockIdx.z;
    const float* W    = (mode == 0) ? Wq : (mode == 1) ? Wk : Wv;
    const float* bias = (mode == 0) ? bq : (mode == 1) ? bk : bv;

    const int rowBase = blockIdx.x * BM;
    const int nBase   = blockIdx.y * BN;

    __shared__ unsigned lA[BM * LSTRIDE];   // 128 rows x 32 k (bf16x2 packed)
    __shared__ unsigned lB[BN * LSTRIDE];   //  64 rows x 32 k

    const int tid  = threadIdx.x;
    const int lane = tid & 31;
    const int w    = tid >> 5;
    const int wm   = w >> 1;          // 0..3 : 32-row group
    const int wn   = w & 1;           // 0..1 : 32-col group
    const int half = lane >> 4;
    const int lm   = lane & 15;

    v8f acc00 = {}, acc01 = {}, acc10 = {}, acc11 = {};

    for (int kb0 = 0; kb0 < D_; kb0 += 32) {
        // cooperative fill: A tile (2048 uints), W tile (1024 uints)
        #pragma unroll
        for (int i = 0; i < 8; ++i) {
            int idx = tid + i * 256;
            int r = idx >> 4, kp = idx & 15;
            const float* g = x + (rowBase + r) * D_ + kb0 + kp * 2;
            lA[r * LSTRIDE + kp] = pack2(g[0], g[1]);
        }
        #pragma unroll
        for (int i = 0; i < 4; ++i) {
            int idx = tid + i * 256;
            int r = idx >> 4, kp = idx & 15;
            const float* g = W + (nBase + r) * D_ + kb0 + kp * 2;
            lB[r * LSTRIDE + kp] = pack2(g[0], g[1]);
        }
        // prefetch next K tile into caches (global_prefetch_b8)
        if (kb0 + 32 < D_) {
            __builtin_prefetch(x + (rowBase + (tid >> 1)) * D_ + kb0 + 32 + (tid & 1) * 16, 0, 3);
            __builtin_prefetch(W + (nBase + (tid >> 2)) * D_ + kb0 + 32 + (tid & 3) * 8, 0, 3);
        }
        __syncthreads();

        FragB16 fa0, fa1, fb0, fb1;
        const int rA0 = wm * 32 + lm, rA1 = rA0 + 16;
        const int nB0 = wn * 32 + lm, nB1 = nB0 + 16;
        #pragma unroll
        for (int v = 0; v < 8; ++v) {
            int ka = afrag_k0(v, half) >> 1;
            fa0.u[v] = lA[rA0 * LSTRIDE + ka];
            fa1.u[v] = lA[rA1 * LSTRIDE + ka];
            int kb = afrag_k0(v, half) >> 1;
            fb0.u[v] = lB[nB0 * LSTRIDE + kb];
            fb1.u[v] = lB[nB1 * LSTRIDE + kb];
        }
        acc00 = __builtin_amdgcn_wmma_f32_16x16x32_bf16(false, fa0.v, false, fb0.v, (short)0, acc00, false, false);
        acc01 = __builtin_amdgcn_wmma_f32_16x16x32_bf16(false, fa0.v, false, fb1.v, (short)0, acc01, false, false);
        acc10 = __builtin_amdgcn_wmma_f32_16x16x32_bf16(false, fa1.v, false, fb0.v, (short)0, acc10, false, false);
        acc11 = __builtin_amdgcn_wmma_f32_16x16x32_bf16(false, fa1.v, false, fb1.v, (short)0, acc11, false, false);
        __syncthreads();
    }

    // epilogue: C layout row = v + 8*half, col = lane&15
    #pragma unroll
    for (int i = 0; i < 2; ++i) {
        #pragma unroll
        for (int j = 0; j < 2; ++j) {
            const v8f acc = (i == 0) ? (j == 0 ? acc00 : acc01) : (j == 0 ? acc10 : acc11);
            #pragma unroll
            for (int v = 0; v < 8; ++v) {
                int row = rowBase + wm * 32 + i * 16 + v + 8 * half;
                int col = nBase + wn * 32 + j * 16 + lm;
                float val = acc[v] + bias[col];
                int b = row >> 11, s = row & (S_ - 1);
                int h = col >> 6,  dh = col & (DH_ - 1);
                unsigned short bfv = f2bf(val);
                if (mode == 0)      qo[(((b * H_ + h) * S_) + s) * DH_ + dh] = bfv;
                else if (mode == 1) ko[(((b * H_ + h) * S_) + s) * DH_ + dh] = bfv;
                else                vo[(((b * H_ + h) * DH_) + dh) * S_ + s] = bfv;  // V^T
            }
        }
    }
}

// ---------------------------------------------------------------------------
// Kernel 2: causal flash attention.  One wave per (head, 16-query tile).
//   grid = (S/16, B*H)  block = 32
//   q,k: [BH,S,DH] bf16   vT: [BH,DH,S] bf16   out: [B,S,D] bf16
// ---------------------------------------------------------------------------
__global__ __launch_bounds__(32) void attn_kernel(
    const unsigned short* __restrict__ q,
    const unsigned short* __restrict__ k,
    const unsigned short* __restrict__ vT,
    unsigned short* __restrict__ out)
{
    const int qbase = blockIdx.x * 16;
    const int head  = blockIdx.y;            // n = b*H + h
    const int lane  = threadIdx.x;
    const int half  = lane >> 4;
    const int lm    = lane & 15;
    const float scale = 0.125f;              // 1/sqrt(64)

    const unsigned short* qh = q  + head * S_ * DH_;
    const unsigned short* kh = k  + head * S_ * DH_;
    const unsigned short* vh = vT + head * DH_ * S_;

    __shared__ unsigned short plds[16][34];  // P tile staging (C-layout -> A-layout)

    // Q A-fragments for both 32-wide K chunks of DH=64 (loaded once)
    FragB16 aq[2];
    #pragma unroll
    for (int d = 0; d < 2; ++d)
        #pragma unroll
        for (int v = 0; v < 8; ++v) {
            int k0 = d * 32 + afrag_k0(v, half);
            aq[d].u[v] = *(const unsigned*)(qh + (qbase + lm) * DH_ + k0);
        }

    v8f oacc[4] = {{}, {}, {}, {}};
    float m_i[8], l_i[8];
    #pragma unroll
    for (int v = 0; v < 8; ++v) { m_i[v] = -INFINITY; l_i[v] = 0.0f; }

    const int nkb = (qbase >> 5) + 1;        // causal: 32-key blocks up to the diagonal
    for (int kb = 0; kb < nkb; ++kb) {
        const int kbase = kb * 32;

        // scores: two 16x16 tiles, each accumulated over DH=64 (2 WMMAs)
        v8f s0 = {}, s1 = {};
        #pragma unroll
        for (int d = 0; d < 2; ++d) {
            FragB16 fb0, fb1;
            #pragma unroll
            for (int v = 0; v < 8; ++v) {
                int k0 = d * 32 + afrag_k0(v, half);
                fb0.u[v] = *(const unsigned*)(kh + (kbase + lm)      * DH_ + k0);
                fb1.u[v] = *(const unsigned*)(kh + (kbase + 16 + lm) * DH_ + k0);
            }
            s0 = __builtin_amdgcn_wmma_f32_16x16x32_bf16(false, aq[d].v, false, fb0.v, (short)0, s0, false, false);
            s1 = __builtin_amdgcn_wmma_f32_16x16x32_bf16(false, aq[d].v, false, fb1.v, (short)0, s1, false, false);
        }

        // online softmax per row (row = v + 8*half; col = lane&15)
        #pragma unroll
        for (int v = 0; v < 8; ++v) {
            int rg = qbase + v + 8 * half;
            float e0 = s0[v] * scale;
            float e1 = s1[v] * scale;
            if (kbase + lm      > rg) e0 = -INFINITY;   // causal mask
            if (kbase + 16 + lm > rg) e1 = -INFINITY;
            float mnew = fmaxf(m_i[v], fmaxf(hmax16(e0), hmax16(e1)));
            float corr = __expf(m_i[v] - mnew);
            float p0 = __expf(e0 - mnew);
            float p1 = __expf(e1 - mnew);
            l_i[v] = l_i[v] * corr + hsum16(p0) + hsum16(p1);
            m_i[v] = mnew;
            #pragma unroll
            for (int t = 0; t < 4; ++t) oacc[t][v] *= corr;
            int r = v + 8 * half;
            plds[r][lm]      = f2bf(p0);
            plds[r][16 + lm] = f2bf(p1);
        }
        __syncthreads();

        // P as A-fragment (16 q x 32 keys)
        FragB16 ap;
        #pragma unroll
        for (int v = 0; v < 8; ++v) {
            int k0 = afrag_k0(v, half);
            ap.u[v] = *(const unsigned*)&plds[lm][k0];
        }
        // O += P @ V   (V^T layout makes B-frag loads contiguous)
        #pragma unroll
        for (int t = 0; t < 4; ++t) {
            FragB16 fv;
            int d = t * 16 + lm;
            #pragma unroll
            for (int v = 0; v < 8; ++v) {
                int k0 = afrag_k0(v, half);
                fv.u[v] = *(const unsigned*)(vh + d * S_ + kbase + k0);
            }
            oacc[t] = __builtin_amdgcn_wmma_f32_16x16x32_bf16(false, ap.v, false, fv.v, (short)0, oacc[t], false, false);
        }
        __syncthreads();
    }

    // normalize and write to concat-head layout [B,S,D]
    const int b = head / H_, h = head % H_;
    #pragma unroll
    for (int v = 0; v < 8; ++v) {
        float rcp = 1.0f / l_i[v];
        int s = qbase + v + 8 * half;
        #pragma unroll
        for (int t = 0; t < 4; ++t) {
            int d = t * 16 + lm;
            out[(b * S_ + s) * D_ + h * DH_ + d] = f2bf(oacc[t][v] * rcp);
        }
    }
}

// ---------------------------------------------------------------------------
// Kernel 3: output projection.  out = attn @ Wo^T + bo  (bf16 in, fp32 out)
//   grid = (M/128, N/64)  block = 256.  A tile uses gfx1250 async->LDS copies.
// ---------------------------------------------------------------------------
__global__ __launch_bounds__(256) void oproj_kernel(
    const unsigned short* __restrict__ A,   // [4096,1024] bf16
    const float* __restrict__ W,            // [1024,1024] fp32
    const float* __restrict__ bias,
    float* __restrict__ out)                // [4096,1024] fp32
{
    const int rowBase = blockIdx.x * BM;
    const int nBase   = blockIdx.y * BN;

    __shared__ unsigned lA[BM * LSTRIDE];
    __shared__ unsigned lB[BN * LSTRIDE];

    const int tid  = threadIdx.x;
    const int lane = tid & 31;
    const int w    = tid >> 5;
    const int wm   = w >> 1;
    const int wn   = w & 1;
    const int half = lane >> 4;
    const int lm   = lane & 15;

    v8f acc00 = {}, acc01 = {}, acc10 = {}, acc11 = {};

    for (int kb0 = 0; kb0 < D_; kb0 += 32) {
#if defined(HAVE_ASYNC_LDS)
        // Async copy A tile: 128 rows x 64B = 512 B128 transfers, 2 per thread.
        // LDS rows are 80B (16B aligned); low 32 bits of a generic LDS pointer
        // are the LDS byte offset (ISA aperture rule).
        #pragma unroll
        for (int i = 0; i < 2; ++i) {
            int slot = tid + i * 256;
            int r = slot >> 2, q4 = slot & 3;
            const unsigned short* g = A + (rowBase + r) * D_ + kb0 + q4 * 8;
            unsigned lds_off = (unsigned)(uintptr_t)(const void*)&lA[r * LSTRIDE + q4 * 4];
            __builtin_amdgcn_global_load_async_to_lds_b128(
                (AS1 v4i_vs*)(uintptr_t)g, (AS3 v4i_vs*)(uintptr_t)lds_off, 0, 0);
        }
#else
        const unsigned* ga = (const unsigned*)A;
        #pragma unroll
        for (int i = 0; i < 8; ++i) {
            int idx = tid + i * 256;
            int r = idx >> 4, kp = idx & 15;
            lA[r * LSTRIDE + kp] = ga[(((rowBase + r) * D_ + kb0) >> 1) + kp];
        }
#endif
        #pragma unroll
        for (int i = 0; i < 4; ++i) {
            int idx = tid + i * 256;
            int r = idx >> 4, kp = idx & 15;
            const float* g = W + (nBase + r) * D_ + kb0 + kp * 2;
            lB[r * LSTRIDE + kp] = pack2(g[0], g[1]);
        }
        if (kb0 + 32 < D_) {
            __builtin_prefetch(A + (rowBase + (tid >> 1)) * D_ + kb0 + 32 + (tid & 1) * 16, 0, 3);
            __builtin_prefetch(W + (nBase + (tid >> 2)) * D_ + kb0 + 32 + (tid & 3) * 8, 0, 3);
        }
        async_wait0();
        __syncthreads();

        FragB16 fa0, fa1, fb0, fb1;
        const int rA0 = wm * 32 + lm, rA1 = rA0 + 16;
        const int nB0 = wn * 32 + lm, nB1 = nB0 + 16;
        #pragma unroll
        for (int v = 0; v < 8; ++v) {
            int ka = afrag_k0(v, half) >> 1;
            fa0.u[v] = lA[rA0 * LSTRIDE + ka];
            fa1.u[v] = lA[rA1 * LSTRIDE + ka];
            int kb = afrag_k0(v, half) >> 1;
            fb0.u[v] = lB[nB0 * LSTRIDE + kb];
            fb1.u[v] = lB[nB1 * LSTRIDE + kb];
        }
        acc00 = __builtin_amdgcn_wmma_f32_16x16x32_bf16(false, fa0.v, false, fb0.v, (short)0, acc00, false, false);
        acc01 = __builtin_amdgcn_wmma_f32_16x16x32_bf16(false, fa0.v, false, fb1.v, (short)0, acc01, false, false);
        acc10 = __builtin_amdgcn_wmma_f32_16x16x32_bf16(false, fa1.v, false, fb0.v, (short)0, acc10, false, false);
        acc11 = __builtin_amdgcn_wmma_f32_16x16x32_bf16(false, fa1.v, false, fb1.v, (short)0, acc11, false, false);
        __syncthreads();
    }

    #pragma unroll
    for (int i = 0; i < 2; ++i) {
        #pragma unroll
        for (int j = 0; j < 2; ++j) {
            const v8f acc = (i == 0) ? (j == 0 ? acc00 : acc01) : (j == 0 ? acc10 : acc11);
            #pragma unroll
            for (int v = 0; v < 8; ++v) {
                int row = rowBase + wm * 32 + i * 16 + v + 8 * half;
                int col = nBase + wn * 32 + j * 16 + lm;
                out[row * D_ + col] = acc[v] + bias[col];
            }
        }
    }
}

// ---------------------------------------------------------------------------
extern "C" void kernel_launch(void* const* d_in, const int* in_sizes, int n_in,
                              void* d_out, int out_size, void* d_ws, size_t ws_size,
                              hipStream_t stream)
{
    (void)in_sizes; (void)n_in; (void)out_size; (void)ws_size;

    const float* x  = (const float*)d_in[0];
    // d_in[1] = mask (bool) — causal tril, computed analytically in-kernel
    const float* Wq = (const float*)d_in[2];
    const float* bq = (const float*)d_in[3];
    const float* Wk = (const float*)d_in[4];
    const float* bk = (const float*)d_in[5];
    const float* Wv = (const float*)d_in[6];
    const float* bv = (const float*)d_in[7];
    const float* Wo = (const float*)d_in[8];
    const float* bo = (const float*)d_in[9];
    float* out = (float*)d_out;

    // workspace layout (bf16): q | k | vT | attn_out, 8 MB each = 32 MB total
    const size_t seg = (size_t)B_ * H_ * S_ * DH_;   // 4M elements
    unsigned short* qb = (unsigned short*)d_ws;
    unsigned short* kb = qb + seg;
    unsigned short* vT = kb + seg;
    unsigned short* ao = vT + seg;

    dim3 g1(MROWS / BM, D_ / BN, 3);
    qkv_kernel<<<g1, 256, 0, stream>>>(x, Wq, Wk, Wv, bq, bk, bv, qb, kb, vT);

    dim3 g2(S_ / 16, B_ * H_);
    attn_kernel<<<g2, 32, 0, stream>>>(qb, kb, vT, ao);

    dim3 g3(MROWS / BM, D_ / BN);
    oproj_kernel<<<g3, 256, 0, stream>>>(ao, Wo, bo, out);
}
